// MyLSTM_12180527252138
// MI455X (gfx1250) — matrix operand order repaired and checked by reading, not compiled, not run on hardware
//
#include <hip/hip_runtime.h>
#include <stdint.h>

// ---------------------------------------------------------------------------
// LSTM inference for MI455X (gfx1250, wave32, WMMA + async-to-LDS copies).
//   tokens[128][80] -> emb lookup (padded K=128, bf16)
//   80 sequential steps:  Z = x_t @ Wx + h @ Wh   (bf16 WMMA, f32 accum)
//                          fused gates -> c, h (f32), h also stored bf16
//   final: sigmoid(h @ fcW + fcb)
// Wh(bf16) = 32 MB -> L2-resident across all 80 steps (192 MB L2), so the
// recurrent GEMM is compute-bound; tiles are staged with
// GLOBAL_LOAD_ASYNC_TO_LDS_B128 (ASYNCcnt) into double-buffered LDS so the
// copy of chunk k+1 overlaps the WMMAs of chunk k.
// ---------------------------------------------------------------------------

typedef __bf16 bf16_t;
typedef bf16_t v16bf __attribute__((ext_vector_type(16)));
typedef float  v8f   __attribute__((ext_vector_type(8)));

#define BATCH 128
#define SEQ   80
#define EMBD  100
#define UNITS 2048
#define GDIM  (4 * UNITS)   // 8192 gate columns
#define KPAD  128           // EMB padded to WMMA-friendly 128
#define KC    64            // K chunk staged through LDS
#define NCHUNK 34           // 2 chunks (x-phase, K=128) + 32 chunks (h, K=2048)
#define BUFBYTES 24576      // per LDS buffer: A 16KB + B 8KB

union Frag { uint4 u[2]; v16bf v; };

__device__ __forceinline__ unsigned short f2bf(float f) {
    unsigned int u = __float_as_uint(f);
    unsigned int r = (u + 0x7FFFu + ((u >> 16) & 1u)) >> 16;  // RNE
    return (unsigned short)r;
}
__device__ __forceinline__ float sigmf(float x) {
    return 1.0f / (1.0f + __expf(-x));
}
__device__ __forceinline__ float tanh_fast(float x) {
    x = fminf(15.0f, fmaxf(-15.0f, x));
    float e = __expf(2.0f * x);
    return (e - 1.0f) / (e + 1.0f);
}

// ---- one-time weight transforms -------------------------------------------

// WhT[n][k] = bf16(Wh[k][n]);  Wh is [2048][8192] row-major.
__global__ __launch_bounds__(256) void k_transpose_wh(
    const float* __restrict__ Wh, unsigned short* __restrict__ WhT)
{
    __shared__ float tile[32][33];
    const int n0 = blockIdx.x * 32, k0 = blockIdx.y * 32;
    const int tx = threadIdx.x, ty = threadIdx.y;
    #pragma unroll
    for (int i = 0; i < 4; ++i)
        tile[ty + i * 8][tx] = Wh[(k0 + ty + i * 8) * GDIM + n0 + tx];
    __syncthreads();
    #pragma unroll
    for (int i = 0; i < 4; ++i)
        WhT[(size_t)(n0 + ty + i * 8) * UNITS + k0 + tx] = f2bf(tile[tx][ty + i * 8]);
}

// WxT[n][kk] = kk < 100 ? bf16(Wx[kk][n]) : 0;  padded K.
__global__ __launch_bounds__(256) void k_convert_wx(
    const float* __restrict__ Wx, unsigned short* __restrict__ WxT)
{
    int idx = blockIdx.x * 256 + threadIdx.x;      // GDIM*KPAD total
    int n = idx >> 7, kk = idx & 127;
    WxT[idx] = (kk < EMBD) ? f2bf(Wx[kk * GDIM + n]) : (unsigned short)0;
}

// xAll[t][b][kk] = bf16(emb[tokens[b][t]][kk]) (zero-padded to KPAD)
__global__ __launch_bounds__(256) void k_embed(
    const int* __restrict__ tokens, const float* __restrict__ emb,
    unsigned short* __restrict__ xAll)
{
    int idx = blockIdx.x * 256 + threadIdx.x;      // SEQ*BATCH*KPAD total
    int kk = idx & 127;
    int b  = (idx >> 7) & 127;
    int t  = idx >> 14;
    int tok = tokens[b * SEQ + t];
    xAll[idx] = (kk < EMBD) ? f2bf(emb[tok * EMBD + kk]) : (unsigned short)0;
}

__global__ __launch_bounds__(256) void k_init(
    unsigned short* __restrict__ hB0, float* __restrict__ cF)
{
    int idx = blockIdx.x * 256 + threadIdx.x;      // BATCH*UNITS total
    hB0[idx] = 0;
    cF[idx]  = 0.0f;
}

// ---- per-timestep fused GEMM + gate kernel --------------------------------
// grid = 128 blocks, block owns unit strip u0..u0+15 and all 4 gates (64 cols).
// 8 waves: wave w -> rows (w&3)*32..+31, cols (w>>2)*32..+31  (4 C tiles).
// K is streamed in 64-wide chunks via async-to-LDS, double buffered.
__global__ __launch_bounds__(256) void k_step(
    const unsigned short* __restrict__ WhT,   // [8192][2048] bf16
    const unsigned short* __restrict__ WxT,   // [8192][128]  bf16
    const unsigned short* __restrict__ xT,    // [128][128]   bf16 (this t)
    const float* __restrict__ bias,           // [8192]
    const unsigned short* __restrict__ hB_in, // [128][2048] bf16 (read)
    unsigned short* __restrict__ hB_out,      // [128][2048] bf16 (write)
    float* __restrict__ hF,                   // [128][2048] f32
    float* __restrict__ cF)                   // [128][2048] f32
{
    // buf i at i*24KB: A[128][KC] (16KB) then B[64][KC] (8KB).  Zs[128][64]
    // (32KB) overlays bytes 0..32K, used only after the final barrier.
    __shared__ __align__(16) unsigned char smem[2 * BUFBYTES];
    float* Zs = (float*)smem;

    const int tid  = threadIdx.x;
    const int lane = tid & 31;
    const int w    = tid >> 5;
    const int u0   = blockIdx.x * 16;

    const int mbase = (w & 3) * 32;
    const int nbase = (w >> 2) * 32;
    const int l15   = lane & 15;
    const int hi    = lane >> 4;

    v8f acc[2][2];
    #pragma unroll
    for (int i = 0; i < 2; ++i)
        #pragma unroll
        for (int j = 0; j < 2; ++j) acc[i][j] = (v8f)(0.0f);

    // Issue one chunk's async copies: 6 wave-wide B128 ops per wave
    // (4 for A[128][KC], 2 for B[64][KC]); LDS dest = workgroup-LDS offset.
    auto issue = [&](int ci, int buf) {
        const unsigned short* Asrc;
        const unsigned short* Bsrc;
        int pitch, kc;
        if (ci < 2) { Asrc = xT;    Bsrc = WxT; pitch = KPAD;  kc = ci * KC; }
        else        { Asrc = hB_in; Bsrc = WhT; pitch = UNITS; kc = (ci - 2) * KC; }
        const unsigned base = (unsigned)(buf * BUFBYTES);
        #pragma unroll
        for (int i = 0; i < 4; ++i) {                      // A: 1024 segs / block
            int s = tid + i * 256;
            int row = s >> 3, sp = s & 7;
            unsigned lds = base + (unsigned)(row * KC + sp * 8) * 2u;
            const unsigned short* g = Asrc + (size_t)row * pitch + kc + sp * 8;
            asm volatile("global_load_async_to_lds_b128 %0, %1, off"
                         :: "v"(lds), "v"((unsigned long long)(uintptr_t)g)
                         : "memory");
        }
        #pragma unroll
        for (int i = 0; i < 2; ++i) {                      // B: 512 segs / block
            int s = tid + i * 256;
            int nn = s >> 3, sp = s & 7;
            int ng = (nn >> 4) * UNITS + u0 + (nn & 15);   // gate-interleaved cols
            unsigned lds = base + 16384u + (unsigned)(nn * KC + sp * 8) * 2u;
            const unsigned short* g = Bsrc + (size_t)ng * pitch + kc + sp * 8;
            asm volatile("global_load_async_to_lds_b128 %0, %1, off"
                         :: "v"(lds), "v"((unsigned long long)(uintptr_t)g)
                         : "memory");
        }
    };

    issue(0, 0);                                           // prologue

    #pragma unroll 1
    for (int ci = 0; ci < NCHUNK; ++ci) {
        if (ci + 1 < NCHUNK) {
            issue(ci + 1, (ci + 1) & 1);                   // overlap copy w/ compute
            // allow the 6 just-issued ops in flight; wait for chunk ci's 6
            asm volatile("s_wait_asynccnt 0x6" ::: "memory");
        } else {
            asm volatile("s_wait_asynccnt 0x0" ::: "memory");
        }
        __syncthreads();                                   // chunk ci visible to all

        const unsigned short* As = (const unsigned short*)(smem + (ci & 1) * BUFBYTES);
        const unsigned short* Bs = As + 128 * KC;

        #pragma unroll
        for (int kk = 0; kk < KC; kk += 32) {
            Frag a[2], b[2];
            // A 16x32 bf16 frag: lanes<16 hold K {0..7,16..23}, lanes>=16 {8..15,24..31}
            #pragma unroll
            for (int mt = 0; mt < 2; ++mt) {
                const unsigned short* p = As + (mbase + mt * 16 + l15) * KC + kk + hi * 8;
                a[mt].u[0] = *(const uint4*)p;
                a[mt].u[1] = *(const uint4*)(p + 16);
            }
            // B 32x16 bf16 frag: lanes<16 hold K {0..7,16..23} of col n, lanes>=16 {8..15,24..31}
            #pragma unroll
            for (int nt = 0; nt < 2; ++nt) {
                const unsigned short* p = Bs + (nbase + nt * 16 + l15) * KC + kk + hi * 8;
                b[nt].u[0] = *(const uint4*)p;
                b[nt].u[1] = *(const uint4*)(p + 16);
            }
            #pragma unroll
            for (int mt = 0; mt < 2; ++mt)
                #pragma unroll
                for (int nt = 0; nt < 2; ++nt)
                    acc[mt][nt] = __builtin_amdgcn_wmma_f32_16x16x32_bf16(
                        false, a[mt].v, false, b[nt].v,
                        (short)0, acc[mt][nt], false, false);
        }
        __syncthreads();                                   // buffer may be refilled
    }

    // spill accumulators to Zs[128][64] (C tile: vgpr r, lanes<16 -> M=r, N=lane;
    // lanes>=16 -> M=8+r, N=lane-16)
    #pragma unroll
    for (int mt = 0; mt < 2; ++mt)
        #pragma unroll
        for (int nt = 0; nt < 2; ++nt)
            #pragma unroll
            for (int r = 0; r < 8; ++r) {
                int m = mbase + mt * 16 + hi * 8 + r;
                int n = nbase + nt * 16 + l15;
                Zs[m * 64 + n] = acc[mt][nt][r];
            }
    __syncthreads();

    // fused LSTM gate update: 128 batch x 16 units = 2048 cells, 8/thread
    #pragma unroll
    for (int i = 0; i < 8; ++i) {
        int cidx = tid + i * 256;
        int j  = cidx & 15;
        int bb = cidx >> 4;
        int u  = u0 + j;
        float zi = Zs[bb * 64 + j]      + bias[u];
        float zf = Zs[bb * 64 + 16 + j] + bias[UNITS + u];
        float zg = Zs[bb * 64 + 32 + j] + bias[2 * UNITS + u];
        float zo = Zs[bb * 64 + 48 + j] + bias[3 * UNITS + u];
        float cold = cF[bb * UNITS + u];
        float cnew = sigmf(zf) * cold + sigmf(zi) * tanh_fast(zg);
        float hnew = sigmf(zo) * tanh_fast(cnew);
        cF[bb * UNITS + u]     = cnew;
        hF[bb * UNITS + u]     = hnew;
        hB_out[bb * UNITS + u] = f2bf(hnew);
    }
}

// ---- final Dense(1) + sigmoid ---------------------------------------------
__global__ __launch_bounds__(256) void k_final(
    const float* __restrict__ hF, const float* __restrict__ fcW,
    const float* __restrict__ fcb, float* __restrict__ out)
{
    __shared__ float red[256];
    const int b = blockIdx.x, tid = threadIdx.x;
    float s = 0.0f;
    for (int k = tid; k < UNITS; k += 256) s += hF[b * UNITS + k] * fcW[k];
    red[tid] = s;
    __syncthreads();
    for (int off = 128; off > 0; off >>= 1) {
        if (tid < off) red[tid] += red[tid + off];
        __syncthreads();
    }
    if (tid == 0) out[b] = 1.0f / (1.0f + __expf(-(red[0] + fcb[0])));
}

// ---------------------------------------------------------------------------
extern "C" void kernel_launch(void* const* d_in, const int* in_sizes, int n_in,
                              void* d_out, int out_size, void* d_ws, size_t ws_size,
                              hipStream_t stream)
{
    const int*   tokens = (const int*)  d_in[0];
    const float* emb    = (const float*)d_in[1];
    const float* Wx     = (const float*)d_in[2];
    const float* Wh     = (const float*)d_in[3];
    const float* bias   = (const float*)d_in[4];
    const float* fcW    = (const float*)d_in[5];
    const float* fcb    = (const float*)d_in[6];
    float* out = (float*)d_out;

    // workspace layout (bytes), ~41.4 MB total
    char* ws = (char*)d_ws;
    unsigned short* WhT  = (unsigned short*)(ws);                 // 33,554,432
    unsigned short* WxT  = (unsigned short*)(ws + 33554432);      //  2,097,152
    unsigned short* xAll = (unsigned short*)(ws + 35651584);      //  2,621,440
    unsigned short* hB0  = (unsigned short*)(ws + 38273024);      //    524,288
    unsigned short* hB1  = (unsigned short*)(ws + 38797312);      //    524,288
    float*          hF   = (float*)        (ws + 39321600);       //  1,048,576
    float*          cF   = (float*)        (ws + 40370176);       //  1,048,576

    k_transpose_wh<<<dim3(GDIM / 32, UNITS / 32), dim3(32, 8), 0, stream>>>(Wh, WhT);
    k_convert_wx<<<(GDIM * KPAD) / 256, 256, 0, stream>>>(Wx, WxT);
    k_embed<<<(SEQ * BATCH * KPAD) / 256, 256, 0, stream>>>(tokens, emb, xAll);
    k_init<<<(BATCH * UNITS) / 256, 256, 0, stream>>>(hB0, cF);

    for (int t = 0; t < SEQ; ++t) {
        unsigned short* hin  = (t & 1) ? hB1 : hB0;
        unsigned short* hout = (t & 1) ? hB0 : hB1;
        k_step<<<UNITS / 16, 256, 0, stream>>>(
            WhT, WxT, xAll + (size_t)t * BATCH * KPAD, bias, hin, hout, hF, cF);
    }
    k_final<<<BATCH, 256, 0, stream>>>(hF, fcW, fcb, out);
}
